// TransformerDecoders_14946486190704
// MI455X (gfx1250) — matrix operand order repaired and checked by reading, not compiled, not run on hardware
//
#include <hip/hip_runtime.h>
#include <hip/hip_bf16.h>

#if defined(__has_builtin)
#  if __has_builtin(__builtin_amdgcn_global_load_async_to_lds_b128) && \
      __has_builtin(__builtin_amdgcn_s_wait_asynccnt)
#    define USE_ASYNC_LDS 1
#  endif
#endif
#ifndef USE_ASYNC_LDS
#  define USE_ASYNC_LDS 0
#endif

typedef unsigned short u16;
typedef __attribute__((ext_vector_type(16))) __bf16         v16bf;
typedef __attribute__((ext_vector_type(8)))  float          v8f;
typedef __attribute__((ext_vector_type(4)))  unsigned int   v4u;
typedef __attribute__((ext_vector_type(4)))  float          v4f;
typedef __attribute__((ext_vector_type(4)))  unsigned short v4s;

#if USE_ASYNC_LDS
// exact pointee type from the builtin prototype: GCC-style v4i vector
typedef int b128i __attribute__((vector_size(4 * sizeof(int))));
typedef __attribute__((address_space(1))) b128i* gptr_b128;
typedef __attribute__((address_space(3))) b128i* lptr_b128;
#endif

union FragAB { v16bf v; v4u u[2]; u16 s[16]; };

__device__ __forceinline__ v8f wmma_bf16(const FragAB& a, const FragAB& b, v8f c) {
  return __builtin_amdgcn_wmma_f32_16x16x32_bf16(false, a.v, false, b.v,
                                                 (short)0, c, false, false);
}

__device__ __forceinline__ u16 f2bf(float f) {
  unsigned int u = __float_as_uint(f);
  u += 0x7FFFu + ((u >> 16) & 1u);   // round-to-nearest-even
  return (u16)(u >> 16);
}

// ---------------------------------------------------------------- fp32 -> bf16
__global__ void cvt_f32_bf16(const float* __restrict__ src, u16* __restrict__ dst, int n4) {
  int i = blockIdx.x * blockDim.x + threadIdx.x;
  if (i >= n4) return;
  v4f f = *(const v4f*)(src + (size_t)i * 4);
  v4s o = { f2bf(f.x), f2bf(f.y), f2bf(f.z), f2bf(f.w) };
  *(v4s*)(dst + (size_t)i * 4) = o;
}

// ---------------------------------------------------------------- GEMM (bf16 WMMA)
// C[M,N] = A[M,K] @ W[K,N] + bias[N]; compile-time relu / bf16-vs-f32 output.
// Workgroup tile 128x128, 8 waves (4x2), wave tile 32x64.
// K staged 64 at a time through LDS; global loads for the next stage are
// register-double-buffered under the 16 WMMAs of the current stage, and the
// stage after that is touched with global_prefetch.
template<int OUTBF, int RELU>
__global__ __launch_bounds__(256) void gemm_bf16_kernel(
    const u16* __restrict__ A, const u16* __restrict__ W,
    const float* __restrict__ bias,
    float* __restrict__ Cf, u16* __restrict__ Cb,
    int M, int N, int K)
{
  __shared__ u16 As[128][72];   // [m][k], row stride 144B (16B aligned)
  __shared__ u16 Bs[128][72];   // [n][k] (transposed on store)

  const int tid  = threadIdx.x;
  const int lane = tid & 31;
  const int wave = tid >> 5;
  const int wm   = wave >> 1;          // 0..3
  const int wn   = wave & 1;           // 0..1
  const int li   = lane & 15;
  const int lh   = lane >> 4;
  const long blockM = (long)blockIdx.y * 128;
  const long blockN = (long)blockIdx.x * 128;

  v8f zero = {};
  v8f acc[2][4];
#pragma unroll
  for (int i = 0; i < 2; ++i)
#pragma unroll
    for (int j = 0; j < 4; ++j) acc[i][j] = zero;

  const int arow = tid >> 1;           // 0..127
  const int acol = (tid & 1) * 32;     // 0 or 32
  const int bk   = tid >> 2;           // 0..63
  const int bn   = (tid & 3) * 32;     // 0,32,64,96

  const u16* aBase = A + (size_t)(blockM + arow) * K + acol;
  const u16* wBase = W + (size_t)bk * N + blockN + bn;

  v4u aR[4], bR[4];
  {
#pragma unroll
    for (int i = 0; i < 4; ++i) aR[i] = *(const v4u*)(aBase + i * 8);
#pragma unroll
    for (int i = 0; i < 4; ++i) bR[i] = *(const v4u*)(wBase + i * 8);
  }

  for (int k0 = 0; k0 < K; k0 += 64) {
    __syncthreads();
    // stage registers -> LDS
#pragma unroll
    for (int i = 0; i < 4; ++i) *(v4u*)&As[arow][acol + i * 8] = aR[i];
    {
      union { v4u u[4]; u16 s[32]; } tb;
#pragma unroll
      for (int i = 0; i < 4; ++i) tb.u[i] = bR[i];
#pragma unroll
      for (int i = 0; i < 32; ++i) Bs[bn + i][bk] = tb.s[i];
    }
    __syncthreads();

    // global loads for next stage (latency hidden under WMMAs below)
    if (k0 + 64 < K) {
      const u16* ap = aBase + (k0 + 64);
      const u16* wp = wBase + (size_t)(k0 + 64) * N;
#pragma unroll
      for (int i = 0; i < 4; ++i) aR[i] = *(const v4u*)(ap + i * 8);
#pragma unroll
      for (int i = 0; i < 4; ++i) bR[i] = *(const v4u*)(wp + i * 8);
    }
    // prefetch the stage after that into cache
    if (k0 + 128 < K) {
      __builtin_prefetch(aBase + (k0 + 128), 0, 3);
      __builtin_prefetch(wBase + (size_t)(k0 + 128) * N, 0, 3);
    }

#pragma unroll
    for (int ks = 0; ks < 2; ++ks) {
      FragAB af[2], bf[4];
#pragma unroll
      for (int fm = 0; fm < 2; ++fm) {
        const u16* p = &As[wm * 32 + fm * 16 + li][ks * 32 + lh * 8];
        af[fm].u[0] = *(const v4u*)p;
        af[fm].u[1] = *(const v4u*)(p + 16);
      }
#pragma unroll
      for (int fn = 0; fn < 4; ++fn) {
        const u16* p = &Bs[wn * 64 + fn * 16 + li][ks * 32 + lh * 8];
        bf[fn].u[0] = *(const v4u*)p;
        bf[fn].u[1] = *(const v4u*)(p + 16);
      }
#pragma unroll
      for (int fm = 0; fm < 2; ++fm)
#pragma unroll
        for (int fn = 0; fn < 4; ++fn)
          acc[fm][fn] = wmma_bf16(af[fm], bf[fn], acc[fm][fn]);
    }
  }

  // epilogue: bias (+relu), compile-time output format -> straight-line code
#pragma unroll
  for (int fn = 0; fn < 4; ++fn) {
    const long gn = blockN + wn * 64 + fn * 16 + li;
    const float bv = bias[gn];
#pragma unroll
    for (int fm = 0; fm < 2; ++fm) {
      const long gm0 = blockM + wm * 32 + fm * 16 + lh * 8;
#pragma unroll
      for (int r = 0; r < 8; ++r) {
        float v = acc[fm][fn][r] + bv;
        if (RELU) v = fmaxf(v, 0.0f);
        if (OUTBF) Cb[(size_t)(gm0 + r) * N + gn] = f2bf(v);
        else       Cf[(size_t)(gm0 + r) * N + gn] = v;
      }
    }
  }
}

// ---------------------------------------------------------------- flash attention
// One workgroup per (b,h,q-block of 128). 8 waves; wave owns 16 query rows.
// Q,K,V,Z are bf16 in (B,S,E) layout; head h occupies columns [h*64, h*64+64).
__global__ __launch_bounds__(256) void attn_kernel(
    const u16* __restrict__ Qb, const u16* __restrict__ Kb,
    const u16* __restrict__ Vb, u16* __restrict__ Zb,
    int S, int causal)
{
  const int E = 1024, D = 64;
  __shared__ u16 Ks[32][72];       // [key][d]
  __shared__ u16 Vs[64][40];       // [d][key] (transposed)
  __shared__ u16 Ps[8][16][40];    // per-wave P tile [qrow][key]

  const int tid  = threadIdx.x;
  const int lane = tid & 31;
  const int wave = tid >> 5;
  const int li   = lane & 15;
  const int lh   = lane >> 4;
  const int bh   = blockIdx.y;
  const int b    = bh >> 4;
  const int h    = bh & 15;
  const int qblock = blockIdx.x * 128;
  const int q0     = qblock + wave * 16;

  // Q fragments (A layout), loaded once per wave
  FragAB qf[2];
  {
    const u16* qrow = Qb + ((size_t)(b * S + q0 + li) * E + h * D);
#pragma unroll
    for (int ks = 0; ks < 2; ++ks) {
      const u16* p = qrow + ks * 32 + lh * 8;
      qf[ks].u[0] = *(const v4u*)p;
      qf[ks].u[1] = *(const v4u*)(p + 16);
    }
  }

  v8f zero = {};
  v8f o[4];
#pragma unroll
  for (int d = 0; d < 4; ++d) o[d] = zero;
  float m[8], l[8];
#pragma unroll
  for (int r = 0; r < 8; ++r) { m[r] = -1e30f; l[r] = 0.0f; }

  const int lk_key = tid >> 3;         // 0..31
  const int lk_d   = (tid & 7) * 8;    // 0..56

  int jend = causal ? (qblock + 128) : S;
  if (jend > S) jend = S;

  for (int j = 0; j < jend; j += 32) {
    __syncthreads();
    {
      const size_t rowoff = (size_t)(b * S + j + lk_key) * E + h * D + lk_d;
      // K tile: straight per-lane 16B global -> LDS copy.
#if USE_ASYNC_LDS
      __builtin_amdgcn_global_load_async_to_lds_b128(
          (gptr_b128)(Kb + rowoff),
          (lptr_b128)&Ks[lk_key][lk_d], 0, 0);
#else
      v4u kv = *(const v4u*)(Kb + rowoff);
      *(v4u*)&Ks[lk_key][lk_d] = kv;
#endif
      // V tile: transposed store (manual)
      union { v4u u; u16 s[8]; } tv;
      tv.u = *(const v4u*)(Vb + rowoff);
#pragma unroll
      for (int i = 0; i < 8; ++i) Vs[lk_d + i][lk_key] = tv.s[i];
    }
#if USE_ASYNC_LDS
    __builtin_amdgcn_s_wait_asynccnt(0);
#endif
    __syncthreads();

    // scores for two 16-key fragments
    float sv[2][8];
#pragma unroll
    for (int nf = 0; nf < 2; ++nf) {
      FragAB kf0, kf1;
      const u16* kp0 = &Ks[nf * 16 + li][lh * 8];
      kf0.u[0] = *(const v4u*)kp0;
      kf0.u[1] = *(const v4u*)(kp0 + 16);
      const u16* kp1 = &Ks[nf * 16 + li][32 + lh * 8];
      kf1.u[0] = *(const v4u*)kp1;
      kf1.u[1] = *(const v4u*)(kp1 + 16);
      v8f c = zero;
      c = wmma_bf16(qf[0], kf0, c);
      c = wmma_bf16(qf[1], kf1, c);
      const int kcol = j + nf * 16 + li;
#pragma unroll
      for (int r = 0; r < 8; ++r) {
        float s = c[r] * 0.125f;                    // 1/sqrt(64)
        if (causal && (kcol > q0 + lh * 8 + r)) s = -1e9f;
        sv[nf][r] = s;
      }
    }

    // online softmax per row (16 lanes share a row; reduce with xor masks 1..8)
#pragma unroll
    for (int r = 0; r < 8; ++r) {
      float cm = fmaxf(sv[0][r], sv[1][r]);
#pragma unroll
      for (int msk = 8; msk >= 1; msk >>= 1)
        cm = fmaxf(cm, __shfl_xor(cm, msk, 32));
      float mnew = fmaxf(m[r], cm);
      float corr = __expf(m[r] - mnew);
      float p0 = __expf(sv[0][r] - mnew);
      float p1 = __expf(sv[1][r] - mnew);
      float rs = p0 + p1;
#pragma unroll
      for (int msk = 8; msk >= 1; msk >>= 1)
        rs += __shfl_xor(rs, msk, 32);
      l[r] = l[r] * corr + rs;
      m[r] = mnew;
#pragma unroll
      for (int d = 0; d < 4; ++d) o[d][r] *= corr;
      Ps[wave][lh * 8 + r][li]      = f2bf(p0);
      Ps[wave][lh * 8 + r][16 + li] = f2bf(p1);
    }

    // P (C layout) -> A layout via per-wave LDS (same-wave DS ops are in-order)
    FragAB pf;
    {
      const u16* pp = &Ps[wave][li][lh * 8];
      pf.u[0] = *(const v4u*)pp;
      pf.u[1] = *(const v4u*)(pp + 16);
    }
#pragma unroll
    for (int d = 0; d < 4; ++d) {
      FragAB vf;
      const u16* vp = &Vs[d * 16 + li][lh * 8];
      vf.u[0] = *(const v4u*)vp;
      vf.u[1] = *(const v4u*)(vp + 16);
      o[d] = wmma_bf16(pf, vf, o[d]);
    }
  }

  // normalize and store bf16 (B,S,E) with head offset
#pragma unroll
  for (int r = 0; r < 8; ++r) {
    const float inv = 1.0f / l[r];
    const size_t row = (size_t)(b * S + q0 + lh * 8 + r) * E + h * D;
#pragma unroll
    for (int d = 0; d < 4; ++d)
      Zb[row + d * 16 + li] = f2bf(o[d][r] * inv);
  }
}

// ---------------------------------------------------------------- fused add + LayerNorm
// y = LN(a + x) * g + b ; writes fp32 (for residual) and bf16 (for next GEMM)
__global__ __launch_bounds__(256) void add_ln_kernel(
    const float* __restrict__ a, const float* __restrict__ x,
    const float* __restrict__ g, const float* __restrict__ bta,
    float* __restrict__ yf, u16* __restrict__ yb)
{
  const int E = 1024;
  const int row = blockIdx.x;
  const int tid = threadIdx.x;
  const size_t base = (size_t)row * E + (size_t)tid * 4;

  v4f av = *(const v4f*)(a + base);
  v4f xv = *(const v4f*)(x + base);
  float v0 = av.x + xv.x, v1 = av.y + xv.y, v2 = av.z + xv.z, v3 = av.w + xv.w;
  float s = v0 + v1 + v2 + v3;
  float q = v0 * v0 + v1 * v1 + v2 * v2 + v3 * v3;
#pragma unroll
  for (int msk = 16; msk >= 1; msk >>= 1) {
    s += __shfl_xor(s, msk, 32);
    q += __shfl_xor(q, msk, 32);
  }
  __shared__ float rs[8], rq[8];
  const int wave = tid >> 5, lane = tid & 31;
  if (lane == 0) { rs[wave] = s; rq[wave] = q; }
  __syncthreads();
  s = 0.0f; q = 0.0f;
#pragma unroll
  for (int i = 0; i < 8; ++i) { s += rs[i]; q += rq[i]; }
  const float mean = s * (1.0f / 1024.0f);
  const float var  = q * (1.0f / 1024.0f) - mean * mean;
  const float rstd = rsqrtf(var + 1e-3f);

  const int c = tid * 4;
  float o0 = (v0 - mean) * rstd * g[c + 0] + bta[c + 0];
  float o1 = (v1 - mean) * rstd * g[c + 1] + bta[c + 1];
  float o2 = (v2 - mean) * rstd * g[c + 2] + bta[c + 2];
  float o3 = (v3 - mean) * rstd * g[c + 3] + bta[c + 3];
  v4f of = { o0, o1, o2, o3 };
  *(v4f*)(yf + base) = of;
  v4s ob = { f2bf(o0), f2bf(o1), f2bf(o2), f2bf(o3) };
  *(v4s*)(yb + base) = ob;
}

// ---------------------------------------------------------------- host orchestration
extern "C" void kernel_launch(void* const* d_in, const int* in_sizes, int n_in,
                              void* d_out, int out_size, void* d_ws, size_t ws_size,
                              hipStream_t stream)
{
  (void)in_sizes; (void)n_in; (void)out_size; (void)ws_size;
  const int L = 6, E = 1024, Ff = 4096, B = 4, S = 1024;
  const int M = B * S;  // 4096 rows

  const float* x    = (const float*)d_in[0];
  const float* enck = (const float*)d_in[1];
  const float* encv = (const float*)d_in[2];
  const float* sWq = (const float*)d_in[3],  *sWk = (const float*)d_in[4];
  const float* sWv = (const float*)d_in[5],  *sWo = (const float*)d_in[6];
  const float* sbq = (const float*)d_in[7],  *sbk = (const float*)d_in[8];
  const float* sbv = (const float*)d_in[9],  *sbo = (const float*)d_in[10];
  const float* cWq = (const float*)d_in[11], *cWk = (const float*)d_in[12];
  const float* cWv = (const float*)d_in[13], *cWo = (const float*)d_in[14];
  const float* cbq = (const float*)d_in[15], *cbk = (const float*)d_in[16];
  const float* cbv = (const float*)d_in[17], *cbo = (const float*)d_in[18];
  const float* g1 = (const float*)d_in[19], *b1 = (const float*)d_in[20];
  const float* g2 = (const float*)d_in[21], *b2 = (const float*)d_in[22];
  const float* g3 = (const float*)d_in[23], *b3 = (const float*)d_in[24];
  const float* W1  = (const float*)d_in[25], *bf1 = (const float*)d_in[26];
  const float* W2  = (const float*)d_in[27], *bf2 = (const float*)d_in[28];

  // workspace carving
  char* w = (char*)d_ws;
  auto carve = [&](size_t bytes) -> char* {
    char* p = w; w += (bytes + 255) & ~(size_t)255; return p;
  };
  const size_t actF = (size_t)M * E * sizeof(float);   // 16 MiB
  const size_t actB = (size_t)M * E * sizeof(u16);     //  8 MiB
  float* F0  = (float*)carve(actF);
  float* F1  = (float*)carve(actF);
  float* F2  = (float*)carve(actF);
  u16* act = (u16*)carve(actB);
  u16* qb  = (u16*)carve(actB);
  u16* kb  = (u16*)carve(actB);
  u16* vb  = (u16*)carve(actB);
  u16* zb  = (u16*)carve(actB);
  u16* ekb = (u16*)carve(actB);
  u16* evb = (u16*)carve(actB);
  u16* hb  = (u16*)carve((size_t)M * Ff * sizeof(u16)); // 32 MiB
  u16* wb  = (u16*)carve((size_t)E * Ff * sizeof(u16)); //  8 MiB (max weight slab)

  auto cvt = [&](const float* src, u16* dst, size_t n) {
    int n4 = (int)(n / 4);
    cvt_f32_bf16<<<dim3((n4 + 255) / 256), dim3(256), 0, stream>>>(src, dst, n4);
  };
  auto gemm = [&](const u16* Ain, const float* Wf, size_t wElems, const float* bias,
                  float* Cf, u16* Cb, int N, int K, int relu) {
    cvt(Wf, wb, wElems);
    dim3 grid(N / 128, M / 128), blk(256);
    if (Cb) {
      if (relu)
        gemm_bf16_kernel<1, 1><<<grid, blk, 0, stream>>>(Ain, wb, bias, nullptr, Cb, M, N, K);
      else
        gemm_bf16_kernel<1, 0><<<grid, blk, 0, stream>>>(Ain, wb, bias, nullptr, Cb, M, N, K);
    } else {
      gemm_bf16_kernel<0, 0><<<grid, blk, 0, stream>>>(Ain, wb, bias, Cf, nullptr, M, N, K);
    }
  };
  auto attn = [&](int causal) {
    attn_kernel<<<dim3(S / 128, B * 16), dim3(256), 0, stream>>>(
        qb, kb, vb, zb, S, causal);
  };
  auto addln = [&](const float* aa, const float* xx, const float* gg,
                   const float* bb, float* out) {
    add_ln_kernel<<<dim3(M), dim3(256), 0, stream>>>(aa, xx, gg, bb, out, act);
  };

  // init: running x (fp32) + bf16 copies of x, enc_k, enc_v
  (void)hipMemcpyAsync(F0, x, actF, hipMemcpyDeviceToDevice, stream);
  cvt(x,    act, (size_t)M * E);
  cvt(enck, ekb, (size_t)M * E);
  cvt(encv, evb, (size_t)M * E);

  float* cur = F0;    // layer input (fp32)
  float* other = F2;  // spare
  float* t = F1;      // GEMM fp32 output

  for (int l = 0; l < L; ++l) {
    const size_t wo = (size_t)l * E * E, bo = (size_t)l * E;
    // --- masked self-attention ---
    gemm(act, sWq + wo, (size_t)E * E, sbq + bo, nullptr, qb, E, E, 0);
    gemm(act, sWk + wo, (size_t)E * E, sbk + bo, nullptr, kb, E, E, 0);
    gemm(act, sWv + wo, (size_t)E * E, sbv + bo, nullptr, vb, E, E, 0);
    attn(1);
    gemm(zb, sWo + wo, (size_t)E * E, sbo + bo, t, nullptr, E, E, 0);
    addln(t, cur, g1 + bo, b1 + bo, other);            // x1 -> other (+ act bf16)
    // --- cross attention ---
    gemm(act, cWq + wo, (size_t)E * E, cbq + bo, nullptr, qb, E, E, 0);
    gemm(ekb, cWk + wo, (size_t)E * E, cbk + bo, nullptr, kb, E, E, 0);
    gemm(evb, cWv + wo, (size_t)E * E, cbv + bo, nullptr, vb, E, E, 0);
    attn(0);
    gemm(zb, cWo + wo, (size_t)E * E, cbo + bo, t, nullptr, E, E, 0);
    addln(t, other, g2 + bo, b2 + bo, cur);            // x2 -> cur (+ act bf16)
    // --- FFN ---
    gemm(act, W1 + (size_t)l * E * Ff, (size_t)E * Ff, bf1 + (size_t)l * Ff,
         nullptr, hb, Ff, E, 1);                        // relu, bf16 out
    gemm(hb, W2 + (size_t)l * Ff * E, (size_t)Ff * E, bf2 + bo, t, nullptr,
         E, Ff, 0);
    addln(t, cur, g3 + bo, b3 + bo, other);            // x_new -> other
    { float* tmp = cur; cur = other; other = tmp; }
  }

  (void)hipMemcpyAsync(d_out, cur, actF, hipMemcpyDeviceToDevice, stream);
}
